// MultiheadAttention_15839839387655
// MI455X (gfx1250) — matrix operand order repaired and checked by reading, not compiled, not run on hardware
//
#include <hip/hip_runtime.h>
#include <hip/hip_bf16.h>

// ---------------------------------------------------------------------------
// Cosine-similarity multi-head attention for MI455X (gfx1250, wave32, WMMA).
//   S=1024, B=16, D=512, H=8, K=64.
// All matrix math via v_wmma_f32_16x16x32_bf16 (fp32 accumulate).
// Flash-style streaming softmax: no [B,H,S,S] score tensor is materialized.
// Scores are bounded by 1/8 (cosine / sqrt(64)) so no max-subtraction needed.
// Weights are pre-converted to bf16 once; activations are converted inline
// with a biased-round + v_perm_b32 pack. Cross-lane reductions use
// ds_swizzle_b32. Operand loads are staged ahead of WMMA groups so waits
// can retire incrementally instead of fully draining per WMMA.
// ---------------------------------------------------------------------------

typedef __attribute__((ext_vector_type(16))) __bf16 bf16x16;
typedef __attribute__((ext_vector_type(8)))  float  f32x8;

namespace {
constexpr int S  = 1024;   // sequence length
constexpr int NB = 16;     // batch
constexpr int D  = 512;    // model dim
constexpr int H  = 8;      // heads
constexpr int KH = 64;     // head dim
constexpr int HK = H * KH; // 512
constexpr int M  = S * NB; // 16384 rows for the projection GEMMs
}

struct U32x8 { uint4 lo, hi; };

__device__ __forceinline__ bf16x16 make_frag(uint4 lo, uint4 hi) {
  U32x8 u{lo, hi};
  return __builtin_bit_cast(bf16x16, u);
}

// cheap f32 -> bf16 (round-to-nearest, ties away): 2 VALU ops
__device__ __forceinline__ unsigned short f2bfu(float f) {
  unsigned u = __builtin_bit_cast(unsigned, f) + 0x8000u;
  return (unsigned short)(u >> 16);
}

// pack two f32 into one dword of two bf16 (lo = a, hi = b):
// two v_add + one v_perm_b32
__device__ __forceinline__ unsigned pk_bf16x2(float a, float b) {
  unsigned ua = __builtin_bit_cast(unsigned, a) + 0x8000u;
  unsigned ub = __builtin_bit_cast(unsigned, b) + 0x8000u;
  // sel 0-3 -> bytes of src1 (ua), 4-7 -> bytes of src0 (ub)
  return __builtin_amdgcn_perm(ub, ua, 0x07060302u);
}

// build a 16-element bf16 fragment from 16 fp32 (4 float4 chunks)
__device__ __forceinline__ bf16x16 cvt_frag(float4 f0, float4 f1, float4 f2,
                                            float4 f3) {
  uint4 lo, hi;
  lo.x = pk_bf16x2(f0.x, f0.y); lo.y = pk_bf16x2(f0.z, f0.w);
  lo.z = pk_bf16x2(f1.x, f1.y); lo.w = pk_bf16x2(f1.z, f1.w);
  hi.x = pk_bf16x2(f2.x, f2.y); hi.y = pk_bf16x2(f2.z, f2.w);
  hi.z = pk_bf16x2(f3.x, f3.y); hi.w = pk_bf16x2(f3.z, f3.w);
  return make_frag(lo, hi);
}

__device__ __forceinline__ f32x8 wmma_bf16(bf16x16 a, bf16x16 b, f32x8 c) {
  // 8 args: (neg_a, A, neg_b, B, c_mod, C, reuse_a, reuse_b)
  return __builtin_amdgcn_wmma_f32_16x16x32_bf16(false, a, false, b,
                                                 (short)0, c, false, false);
}

// XOR-butterfly sum over each 16-lane half (masks 1,2,4,8 stay in-half).
// ds_swizzle group-of-32 imm: [14:10]=xor, [9:5]=or, [4:0]=and(=0x1F).
__device__ __forceinline__ float reduce16(float x) {
  x += __builtin_bit_cast(float, __builtin_amdgcn_ds_swizzle(
           __builtin_bit_cast(int, x), 0x041f));   // xor 1
  x += __builtin_bit_cast(float, __builtin_amdgcn_ds_swizzle(
           __builtin_bit_cast(int, x), 0x081f));   // xor 2
  x += __builtin_bit_cast(float, __builtin_amdgcn_ds_swizzle(
           __builtin_bit_cast(int, x), 0x101f));   // xor 4
  x += __builtin_bit_cast(float, __builtin_amdgcn_ds_swizzle(
           __builtin_bit_cast(int, x), 0x201f));   // xor 8
  return x;
}

// ---------------------------------------------------------------------------
// Kernel 0: one-time fp32 -> bf16 weight conversion (tiny: 2 MB total).
// ---------------------------------------------------------------------------
__global__ void __launch_bounds__(256)
cvt_weights_kernel(const float* __restrict__ in, unsigned short* __restrict__ out,
                   int n)
{
  const int i = blockIdx.x * blockDim.x + threadIdx.x;
  if (i < n) out[i] = f2bfu(in[i]);
}

// ---------------------------------------------------------------------------
// Kernel 1: fused projection GEMM + per-row L2 normalization.
//   proj[m, h*64+f] = sum_d X[m,d] * W[h,f,d],  m = s*B + b
// One wave per (16-row tile x head). 4 accumulators cover all 64 features of
// the head, so the row norm is available in-register after the K loop.
// Q/K stored as bf16 [b*H+h][s][f]; V stored transposed [b*H+h][f][t].
// ---------------------------------------------------------------------------
__global__ void __launch_bounds__(256)
proj_norm_kernel(const float* __restrict__ X, const unsigned short* __restrict__ W,
                 unsigned short* __restrict__ out, int transposeV)
{
  const int lane = threadIdx.x & 31;
  const int l16  = lane & 15;
  const int hi   = lane >> 4;
  const int h    = threadIdx.x >> 5;   // wave id == head id (8 waves/block)
  const int m0   = blockIdx.x * 16;

  f32x8 acc[4] = {};
  const float* xrow = X + (size_t)(m0 + l16) * D;
  const unsigned short* wbase = W + (size_t)(h * KH + l16) * D;

#pragma unroll 2
  for (int ds = 0; ds < D / 32; ++ds) {
    // Stage all loads for this k-step first, then convert, then 4 WMMAs.
    const float* ap = xrow + ds * 32 + (hi ? 8 : 0);
    const float4 x0 = *(const float4*)(ap + 0);
    const float4 x1 = *(const float4*)(ap + 4);
    const float4 x2 = *(const float4*)(ap + 16);
    const float4 x3 = *(const float4*)(ap + 20);
    uint4 bl[4][2];
#pragma unroll
    for (int nt = 0; nt < 4; ++nt) {
      // B fragment: col n = feature nt*16+l16; k = ds*32 + hi*8 + e (e<8), +16 for e>=8
      const uint4* bp =
          (const uint4*)(wbase + (size_t)nt * 16 * D + ds * 32 + hi * 8);
      bl[nt][0] = bp[0];
      bl[nt][1] = bp[2];
    }
    const bf16x16 a = cvt_frag(x0, x1, x2, x3);
#pragma unroll
    for (int nt = 0; nt < 4; ++nt)
      acc[nt] = wmma_bf16(a, make_frag(bl[nt][0], bl[nt][1]), acc[nt]);
  }

  // Per-row sum of squares: features live across the 16-lane group + 4 tiles.
  float inv[8];
#pragma unroll
  for (int r = 0; r < 8; ++r) {
    float sq = acc[0][r] * acc[0][r] + acc[1][r] * acc[1][r] +
               acc[2][r] * acc[2][r] + acc[3][r] * acc[3][r];
    sq = reduce16(sq);
    float n = sqrtf(sq);
    inv[r] = 1.0f / fmaxf(n, 1e-12f);
  }

#pragma unroll
  for (int nt = 0; nt < 4; ++nt) {
#pragma unroll
    for (int r = 0; r < 8; ++r) {
      const int mrow = m0 + r + hi * 8;          // C layout: M = r (+8 hi half)
      const int s = mrow >> 4;                   // m = s*B + b, B = 16
      const int b = mrow & 15;
      const int f = nt * 16 + l16;
      const unsigned short u = f2bfu(acc[nt][r] * inv[r]);
      if (!transposeV)
        out[(((size_t)(b * H + h)) * S + s) * KH + f] = u;
      else
        out[(((size_t)(b * H + h)) * KH + f) * S + s] = u;
    }
  }
}

// ---------------------------------------------------------------------------
// Kernel 2: streaming (flash-style) attention, one wave per 16-query tile.
// Scores bounded by 1/8 => plain exp, running denominator only.
// P (C-layout) -> A-layout conversion via a private per-wave LDS tile
// (LDS ops are in-order within a wave; no barrier required).
// ---------------------------------------------------------------------------
__global__ void __launch_bounds__(256)
attn_kernel(const unsigned short* __restrict__ wq,
            const unsigned short* __restrict__ wk,
            const unsigned short* __restrict__ wvT,
            unsigned short* __restrict__ att)
{
  __shared__ unsigned short pTile[8][16 * 32];   // 8 waves x (16x32 bf16) = 8KB

  const int lane = threadIdx.x & 31;
  const int l16  = lane & 15;
  const int hi   = lane >> 4;
  const int w    = threadIdx.x >> 5;
  const int bh   = blockIdx.y;                   // b*H + h
  const int q0   = (blockIdx.x * 8 + w) * 16;

  const unsigned short* q = wq  + (size_t)bh * S * KH;
  const unsigned short* k = wk  + (size_t)bh * S * KH;
  const unsigned short* v = wvT + (size_t)bh * KH * S;
  unsigned short* pW = pTile[w];

  // Q fragments (A operand), loaded once: head dim 64 = 2 x k32 steps.
  bf16x16 aQ[2];
#pragma unroll
  for (int ds = 0; ds < 2; ++ds) {
    const uint4* p =
        (const uint4*)(q + (size_t)(q0 + l16) * KH + ds * 32 + (hi ? 8 : 0));
    aQ[ds] = make_frag(p[0], p[2]);              // chunks at k and k+16
  }

  f32x8 accO[4] = {};
  float den[8] = {};

  for (int kt = 0; kt < S / 32; ++kt) {
    // Prefetch next key tile (global_prefetch_b8); 32 rows of 128B each.
    if (kt + 1 < S / 32)
      __builtin_prefetch(k + (size_t)((kt + 1) * 32 + lane) * KH, 0, 0);

    // Stage all 4 K-fragments (2 key subtiles x 2 k-steps), then 4 WMMAs.
    uint4 kl[2][2][2];
#pragma unroll
    for (int j = 0; j < 2; ++j) {
      const unsigned short* kb = k + (size_t)(kt * 32 + j * 16 + l16) * KH;
#pragma unroll
      for (int ds = 0; ds < 2; ++ds) {
        const uint4* bp = (const uint4*)(kb + ds * 32 + hi * 8);
        kl[j][ds][0] = bp[0];
        kl[j][ds][1] = bp[2];
      }
    }
#pragma unroll
    for (int j = 0; j < 2; ++j) {                // two 16-key score subtiles
      f32x8 sc = {};
#pragma unroll
      for (int ds = 0; ds < 2; ++ds)
        sc = wmma_bf16(aQ[ds], make_frag(kl[j][ds][0], kl[j][ds][1]), sc);
#pragma unroll
      for (int r = 0; r < 8; ++r) {
        const float pv = __expf(sc[r] * 0.125f); // 1/sqrt(64); |sc|<=1 -> safe
        den[r] += pv;
        pW[(r + hi * 8) * 32 + j * 16 + l16] = f2bfu(pv);
      }
    }
    // Stage the 4 V-fragments while P is written/re-read through LDS.
    uint4 vl[4][2];
#pragma unroll
    for (int nt = 0; nt < 4; ++nt) {
      const uint4* vp =
          (const uint4*)(v + (size_t)(nt * 16 + l16) * S + kt * 32 + hi * 8);
      vl[nt][0] = vp[0];
      vl[nt][1] = vp[2];
    }
    // Re-load P in A layout (row = l16; k chunks per half-wave).
    const uint4* pp = (const uint4*)(pW + l16 * 32 + (hi ? 8 : 0));
    const bf16x16 pf = make_frag(pp[0], pp[2]);
#pragma unroll
    for (int nt = 0; nt < 4; ++nt)               // 4 x 16 output features
      accO[nt] = wmma_bf16(pf, make_frag(vl[nt][0], vl[nt][1]), accO[nt]);
  }

  float inv[8];
#pragma unroll
  for (int r = 0; r < 8; ++r) {
    inv[r] = 1.0f / reduce16(den[r]);
  }

  const int b  = bh / H;
  const int hh = bh % H;
#pragma unroll
  for (int nt = 0; nt < 4; ++nt) {
#pragma unroll
    for (int r = 0; r < 8; ++r) {
      const int srow = q0 + r + hi * 8;
      att[((size_t)srow * NB + b) * HK + hh * KH + nt * 16 + l16] =
          f2bfu(accO[nt][r] * inv[r]);
    }
  }
}

// ---------------------------------------------------------------------------
// Kernel 3: output projection  out[m,d] = sum_j att[m,j] * Wout[d,j]
// 8 waves/block cover all 512 output features of a 16-row tile.
// Both operands are already bf16 -> zero conversion VALU in the loop.
// ---------------------------------------------------------------------------
__global__ void __launch_bounds__(256)
out_proj_kernel(const unsigned short* __restrict__ att,
                const unsigned short* __restrict__ Wout,
                float* __restrict__ out)
{
  const int lane = threadIdx.x & 31;
  const int l16  = lane & 15;
  const int hi   = lane >> 4;
  const int w    = threadIdx.x >> 5;             // 64 output cols per wave
  const int m0   = blockIdx.x * 16;

  f32x8 acc[4] = {};
  const unsigned short* arow = att + (size_t)(m0 + l16) * HK;
  const unsigned short* wbase = Wout + (size_t)(w * 64 + l16) * HK;

#pragma unroll 2
  for (int ds = 0; ds < HK / 32; ++ds) {
    const uint4* ap = (const uint4*)(arow + ds * 32 + (hi ? 8 : 0));
    const uint4 a0 = ap[0];
    const uint4 a1 = ap[2];
    uint4 bl[4][2];
#pragma unroll
    for (int nt = 0; nt < 4; ++nt) {
      const uint4* bp =
          (const uint4*)(wbase + (size_t)nt * 16 * HK + ds * 32 + hi * 8);
      bl[nt][0] = bp[0];
      bl[nt][1] = bp[2];
    }
    const bf16x16 a = make_frag(a0, a1);
#pragma unroll
    for (int nt = 0; nt < 4; ++nt)
      acc[nt] = wmma_bf16(a, make_frag(bl[nt][0], bl[nt][1]), acc[nt]);
  }

#pragma unroll
  for (int nt = 0; nt < 4; ++nt) {
#pragma unroll
    for (int r = 0; r < 8; ++r) {
      const int m = m0 + r + hi * 8;
      out[(size_t)m * D + w * 64 + nt * 16 + l16] = acc[nt][r];
    }
  }
}

// ---------------------------------------------------------------------------
extern "C" void kernel_launch(void* const* d_in, const int* in_sizes, int n_in,
                              void* d_out, int out_size, void* d_ws,
                              size_t ws_size, hipStream_t stream)
{
  const float* q    = (const float*)d_in[0];
  const float* kk   = (const float*)d_in[1];
  const float* v    = (const float*)d_in[2];
  const float* WQ   = (const float*)d_in[3];
  const float* WK   = (const float*)d_in[4];
  const float* WV   = (const float*)d_in[5];
  const float* Wout = (const float*)d_in[6];

  unsigned short* ws = (unsigned short*)d_ws;
  const size_t szQK = (size_t)NB * H * S * KH;   // 8.39M bf16 elems = 16 MB
  const size_t szW  = (size_t)HK * D;            // 262144 elems = 0.5 MB
  unsigned short* wq   = ws;
  unsigned short* wk   = ws + szQK;
  unsigned short* wvT  = ws + 2 * szQK;
  unsigned short* att  = ws + 3 * szQK;
  unsigned short* bWQ  = ws + 4 * szQK;
  unsigned short* bWK  = bWQ + szW;
  unsigned short* bWV  = bWK + szW;
  unsigned short* bWo  = bWV + szW;

  const int nW = (int)szW;
  cvt_weights_kernel<<<dim3((nW + 255) / 256), 256, 0, stream>>>(WQ, bWQ, nW);
  cvt_weights_kernel<<<dim3((nW + 255) / 256), 256, 0, stream>>>(WK, bWK, nW);
  cvt_weights_kernel<<<dim3((nW + 255) / 256), 256, 0, stream>>>(WV, bWV, nW);
  cvt_weights_kernel<<<dim3((nW + 255) / 256), 256, 0, stream>>>(Wout, bWo, nW);

  proj_norm_kernel<<<dim3(M / 16), 256, 0, stream>>>(q,  bWQ, wq,  0);
  proj_norm_kernel<<<dim3(M / 16), 256, 0, stream>>>(kk, bWK, wk,  0);
  proj_norm_kernel<<<dim3(M / 16), 256, 0, stream>>>(v,  bWV, wvT, 1);
  attn_kernel<<<dim3(S / 16 / 8, NB * H), 256, 0, stream>>>(wq, wk, wvT, att);
  out_proj_kernel<<<dim3(M / 16), 256, 0, stream>>>(att, bWo, (float*)d_out);
}
